// VisionMamba_21431886807228
// MI455X (gfx1250) — matrix operand order repaired and checked by reading, not compiled, not run on hardware
//
#include <hip/hip_runtime.h>
#include <math.h>

// ---------------- problem constants ----------------
#define DEPTH    4
#define D_MODEL  384
#define D_INNER  768
#define D_STATE  16
#define DT_RANK  24
#define D_CONV   4
#define PATCH    16
#define BATCH    4
#define HP       14
#define WP       14
#define NCH      3
#define SEQ      (HP*WP*NCH)     // 588
#define NTOK     (BATCH*SEQ)     // 2352
#define IMG      224
#define MTILES   (NTOK/16)       // 147
#define MBLKS    (NTOK/48)       // 49
#define FR       512             // elements per (tile,kstep) fragment block

typedef __bf16 bf16_t;
typedef __attribute__((ext_vector_type(16))) __bf16 v16bf;
typedef __attribute__((ext_vector_type(8)))  float  v8f;

__device__ __forceinline__ float fsilu(float v) { return v / (1.0f + __expf(-v)); }

// Fragment index helpers (CDNA5 ISA 7.12.2, wave32):
//  A (16x32 bf16): lane L -> m = L&15; elem j (VGPR i=j/2) -> k = (i/4)*16 + (L>=16)*8 + (i&3)*2 + (j&1)
//  B (32x16 bf16): lane L -> n = L&15; elem j -> k = (j/8)*16 + (L>=16)*8 + (j&7)
//  C (16x16 f32) : lane L -> n = L&15; VGPR i -> m = (L>=16)*8 + i
__device__ __forceinline__ int ka_of(int j, int hsel) {
    int i = j >> 1;
    return ((i >> 2) << 4) + (hsel << 3) + ((i & 3) << 1) + (j & 1);
}

// ---------------------------------------------------------------------------
// WMMA GEMM on fragment-packed operands.
// Afrag: [Mtiles][Ksteps][32 lanes][16 bf16]   (Mtiles = M/16)
// Bfrag: [Ntiles][Ksteps][32 lanes][16 bf16]   (Ntiles = N/16)
// One wave per block computes a 48(M) x 64(N) macro-tile: 3x4 = 12 WMMA tiles.
// Inner loop: 7 x 32B coalesced loads -> 12 v_wmma_f32_16x16x32_bf16.
// ---------------------------------------------------------------------------
__global__ __launch_bounds__(32) void k_wmma_gemm(
    const bf16_t* __restrict__ Af, const bf16_t* __restrict__ Bf,
    float* __restrict__ C, int Ksteps, int ldc)
{
    const int lane = threadIdx.x;
    const int mA   = lane & 15;
    const int hsel = lane >> 4;
    const bf16_t* Ab = Af + ((size_t)blockIdx.y * 3) * Ksteps * FR + lane * 16;
    const bf16_t* Bb = Bf + ((size_t)blockIdx.x * 4) * Ksteps * FR + lane * 16;

    v8f acc[12];
#pragma unroll
    for (int i = 0; i < 12; ++i) acc[i] = (v8f){};

    for (int s = 0; s < Ksteps; ++s) {
        if (s + 1 < Ksteps) {        // -> global_prefetch_b8
            __builtin_prefetch(Ab + (size_t)(s + 1) * FR, 0, 1);
            __builtin_prefetch(Bb + (size_t)(s + 1) * FR, 0, 1);
        }
        v16bf a[3], b[4];
#pragma unroll
        for (int mi = 0; mi < 3; ++mi)
            a[mi] = *(const v16bf*)(Ab + ((size_t)mi * Ksteps + s) * FR);
#pragma unroll
        for (int ni = 0; ni < 4; ++ni)
            b[ni] = *(const v16bf*)(Bb + ((size_t)ni * Ksteps + s) * FR);
#pragma unroll
        for (int mi = 0; mi < 3; ++mi)
#pragma unroll
            for (int ni = 0; ni < 4; ++ni)
                acc[mi * 4 + ni] = __builtin_amdgcn_wmma_f32_16x16x32_bf16(
                    false, a[mi], false, b[ni], (short)0, acc[mi * 4 + ni],
                    false, false);
    }

    const int m0 = blockIdx.y * 48;
    const int n0 = blockIdx.x * 64;
#pragma unroll
    for (int mi = 0; mi < 3; ++mi)
#pragma unroll
        for (int i = 0; i < 8; ++i) {
            int m = m0 + mi * 16 + (hsel << 3) + i;
            float* crow = C + (size_t)m * ldc + n0 + mA;
#pragma unroll
            for (int ni = 0; ni < 4; ++ni)
                crow[ni * 16] = acc[mi * 4 + ni][i];
        }
}

// -------- stage A (f32 matrix -> bf16 fragment layout, zero padded) --------
__global__ void k_stageA(const float* __restrict__ src, int srcStride,
                         int srcRows, int srcCols,
                         bf16_t* __restrict__ dst, int Ksteps, int total)
{
    int idx = blockIdx.x * blockDim.x + threadIdx.x;
    if (idx >= total) return;
    int j = idx & 15, L = (idx >> 4) & 31, rest = idx >> 9;
    int s = rest % Ksteps, t = rest / Ksteps;
    int m = t * 16 + (L & 15);
    int k = s * 32 + ka_of(j, L >> 4);
    float v = (m < srcRows && k < srcCols) ? src[(size_t)m * srcStride + k] : 0.f;
    dst[idx] = (bf16_t)v;
}

// -------- stage B (f32 [K x N] or transposed [N x K] -> bf16 fragments) ----
__global__ void k_stageB(const float* __restrict__ src, int ldS,
                         int Kdim, int Ndim, int trans,
                         bf16_t* __restrict__ dst, int Ksteps, int total)
{
    int idx = blockIdx.x * blockDim.x + threadIdx.x;
    if (idx >= total) return;
    int j = idx & 15, L = (idx >> 4) & 31, rest = idx >> 9;
    int s = rest % Ksteps, t = rest / Ksteps;
    int n = t * 16 + (L & 15);
    int k = s * 32 + ((j >> 3) << 4) + ((L >> 4) << 3) + (j & 7);
    float v = 0.f;
    if (k < Kdim && n < Ndim)
        v = trans ? src[(size_t)n * ldS + k] : src[(size_t)k * ldS + n];
    dst[idx] = (bf16_t)v;
}

// -------- im2col directly into A-fragment layout (patch embed) -------------
__global__ void k_im2colA(const float* __restrict__ x, bf16_t* __restrict__ dst)
{
    const int Ksteps = 8;                       // K = 256
    int idx = blockIdx.x * blockDim.x + threadIdx.x;
    if (idx >= MTILES * Ksteps * FR) return;
    int j = idx & 15, L = (idx >> 4) & 31, rest = idx >> 9;
    int s = rest % Ksteps, t = rest / Ksteps;
    int m = t * 16 + (L & 15);                  // token row
    int k = s * 32 + ka_of(j, L >> 4);          // p*16+q
    int b  = m / SEQ;
    int tt = m % SEQ;
    int c  = tt % NCH;
    int pw = tt / NCH;
    int hh = pw / WP, ww = pw % WP;
    int p = k >> 4, q = k & 15;
    dst[idx] = (bf16_t)x[(((size_t)b * NCH + c) * IMG + hh * PATCH + p) * IMG
                         + ww * PATCH + q];
}

// -------- gate directly into A-fragment layout: bf16(y * silu(z)) ----------
__global__ void k_gateA(const float* __restrict__ y, const float* __restrict__ xz,
                        bf16_t* __restrict__ dst)
{
    const int Ksteps = 24;                      // K = 768
    int idx = blockIdx.x * blockDim.x + threadIdx.x;
    if (idx >= MTILES * Ksteps * FR) return;
    int j = idx & 15, L = (idx >> 4) & 31, rest = idx >> 9;
    int s = rest % Ksteps, t = rest / Ksteps;
    int row = t * 16 + (L & 15);
    int e   = s * 32 + ka_of(j, L >> 4);
    float z = xz[(size_t)row * (2 * D_INNER) + D_INNER + e];
    dst[idx] = (bf16_t)(y[(size_t)row * D_INNER + e] * fsilu(z));
}

__global__ void k_patch_finish(float* __restrict__ tok,
                               const float* __restrict__ pb,
                               const float* __restrict__ ce)
{
    int idx = blockIdx.x * blockDim.x + threadIdx.x;
    if (idx >= NTOK * D_MODEL) return;
    int e = idx % D_MODEL;
    int r = idx / D_MODEL;
    int c = (r % SEQ) % NCH;
    tok[idx] += pb[e] + ce[c * D_MODEL + e];
}

// -------- residual add + layernorm (also used for the final LN) ------------
__global__ __launch_bounds__(128) void k_resid_ln(
    const float* __restrict__ addend, float* __restrict__ residual,
    float* __restrict__ outp, const float* __restrict__ w,
    const float* __restrict__ b, int accumulate)
{
    __shared__ float red[128];
    __shared__ float s_mean, s_rstd;
    const int row = blockIdx.x;
    const int tid = threadIdx.x;
    float*       rrow = residual + (size_t)row * D_MODEL;
    const float* arow = addend   + (size_t)row * D_MODEL;

    float local[3];
    float sum = 0.f;
#pragma unroll
    for (int j = 0; j < 3; ++j) {
        int c = tid + j * 128;
        float v = arow[c];
        if (accumulate) v += rrow[c];
        rrow[c] = v;
        local[j] = v;
        sum += v;
    }
    red[tid] = sum; __syncthreads();
    for (int s = 64; s > 0; s >>= 1) { if (tid < s) red[tid] += red[tid + s]; __syncthreads(); }
    if (tid == 0) s_mean = red[0] * (1.0f / D_MODEL);
    __syncthreads();
    float mean = s_mean, vs = 0.f;
#pragma unroll
    for (int j = 0; j < 3; ++j) { float d = local[j] - mean; vs += d * d; }
    red[tid] = vs; __syncthreads();
    for (int s = 64; s > 0; s >>= 1) { if (tid < s) red[tid] += red[tid + s]; __syncthreads(); }
    if (tid == 0) s_rstd = rsqrtf(red[0] * (1.0f / D_MODEL) + 1e-5f);
    __syncthreads();
    float rstd = s_rstd;
#pragma unroll
    for (int j = 0; j < 3; ++j) {
        int c = tid + j * 128;
        outp[(size_t)row * D_MODEL + c] = (local[j] - mean) * rstd * w[c] + b[c];
    }
}

// -------- causal depthwise conv (K=4) + SiLU -------------------------------
__global__ void k_conv_silu(const float* __restrict__ xz,
                            const float* __restrict__ cw,
                            const float* __restrict__ cb,
                            float* __restrict__ xc)
{
    int idx = blockIdx.x * blockDim.x + threadIdx.x;
    if (idx >= NTOK * D_INNER) return;
    int c   = idx % D_INNER;
    int row = idx / D_INNER;
    int l   = row % SEQ;
    int b   = row / SEQ;
    float acc = cb[c];
#pragma unroll
    for (int k = 0; k < D_CONV; ++k) {
        int ls = l - (D_CONV - 1) + k;
        if (ls >= 0)
            acc += cw[c * D_CONV + k] *
                   xz[((size_t)b * SEQ + ls) * (2 * D_INNER) + c];
    }
    xc[idx] = fsilu(acc);
}

// -------- softplus(dt + dt_b) ----------------------------------------------
__global__ void k_softplus(float* __restrict__ dt, const float* __restrict__ dtb)
{
    int idx = blockIdx.x * blockDim.x + threadIdx.x;
    if (idx >= NTOK * D_INNER) return;
    float v = dt[idx] + dtb[idx % D_INNER];
    dt[idx] = (v > 20.f) ? v : log1pf(__expf(v));
}

// -------- selective scan: one thread per (batch, channel) ------------------
__global__ __launch_bounds__(256) void k_scan(
    const float* __restrict__ dt,  const float* __restrict__ xc,
    const float* __restrict__ dbc, const float* __restrict__ A_log,
    const float* __restrict__ Dsk, float* __restrict__ y)
{
    int tid = blockIdx.x * blockDim.x + threadIdx.x;
    if (tid >= BATCH * D_INNER) return;
    int b = tid / D_INNER;
    int e = tid % D_INNER;
    float A[D_STATE], h[D_STATE];
#pragma unroll
    for (int s = 0; s < D_STATE; ++s) {
        A[s] = -__expf(A_log[e * D_STATE + s]);
        h[s] = 0.f;
    }
    float dsk = Dsk[e];
    for (int l = 0; l < SEQ; ++l) {
        size_t row = (size_t)b * SEQ + l;
        float d  = dt[row * D_INNER + e];
        float xv = xc[row * D_INNER + e];
        float dx = d * xv;
        const float* brow = dbc + row * 64;
        float acc = 0.f;
#pragma unroll
        for (int s = 0; s < D_STATE; ++s) {
            float dA = __expf(d * A[s]);
            h[s] = dA * h[s] + dx * brow[DT_RANK + s];
            acc += h[s] * brow[DT_RANK + D_STATE + s];
        }
        y[row * D_INNER + e] = acc + xv * dsk;
    }
}

// ===========================================================================
extern "C" void kernel_launch(void* const* d_in, const int* in_sizes, int n_in,
                              void* d_out, int out_size, void* d_ws, size_t ws_size,
                              hipStream_t stream)
{
    (void)in_sizes; (void)n_in; (void)out_size; (void)ws_size;
    const float* x        = (const float*)d_in[0];
    const float* patch_w  = (const float*)d_in[1];
    const float* patch_b  = (const float*)d_in[2];
    const float* chan_emb = (const float*)d_in[3];
    const float* norm_w   = (const float*)d_in[4];
    const float* norm_b   = (const float*)d_in[5];
    const float* in_w     = (const float*)d_in[6];
    const float* conv_w   = (const float*)d_in[7];
    const float* conv_b   = (const float*)d_in[8];
    const float* x_w      = (const float*)d_in[9];
    const float* dt_w     = (const float*)d_in[10];
    const float* dt_b     = (const float*)d_in[11];
    const float* A_log    = (const float*)d_in[12];
    const float* D_skip   = (const float*)d_in[13];
    const float* out_w    = (const float*)d_in[14];
    const float* normf_w  = (const float*)d_in[15];
    const float* normf_b  = (const float*)d_in[16];
    float* out = (float*)d_out;

    // workspace carve-out
    char* p = (char*)d_ws;
    auto alloc = [&](size_t bytes) -> char* {
        char* r = p; p += (bytes + 255) & ~(size_t)255; return r;
    };
    float*  tokens   = (float*)alloc((size_t)NTOK * D_MODEL * 4);
    float*  residual = (float*)alloc((size_t)NTOK * D_MODEL * 4);
    float*  hidden   = (float*)alloc((size_t)NTOK * D_MODEL * 4);
    float*  hs       = (float*)alloc((size_t)NTOK * D_MODEL * 4);
    float*  xz       = (float*)alloc((size_t)NTOK * 2 * D_INNER * 4);
    float*  xc       = (float*)alloc((size_t)NTOK * D_INNER * 4);
    float*  dbc      = (float*)alloc((size_t)NTOK * 64 * 4);
    float*  dtbuf    = (float*)alloc((size_t)NTOK * D_INNER * 4);
    float*  ybuf     = (float*)alloc((size_t)NTOK * D_INNER * 4);
    bf16_t* sA       = (bf16_t*)alloc((size_t)MTILES * 24 * FR * 2);   // up to K=768
    bf16_t* sB       = (bf16_t*)alloc((size_t)96 * 12 * FR * 2);       // up to N=1536,K=384

    const int T = 256;
    auto cd = [](int a, int b) { return (a + b - 1) / b; };

    // ---- patch embedding as WMMA GEMM (2352 x 384 x 256) ----
    {
        const int Ks = 8;
        k_im2colA<<<cd(MTILES * Ks * FR, T), T, 0, stream>>>(x, sA);
        k_stageB<<<cd((D_MODEL/16) * Ks * FR, T), T, 0, stream>>>(
            patch_w, 256, 256, D_MODEL, 1, sB, Ks, (D_MODEL/16) * Ks * FR);
        k_wmma_gemm<<<dim3(D_MODEL / 64, MBLKS), 32, 0, stream>>>(
            sA, sB, tokens, Ks, D_MODEL);
        k_patch_finish<<<cd(NTOK * D_MODEL, T), T, 0, stream>>>(tokens, patch_b, chan_emb);
    }

    // ---- DEPTH mamba layers ----
    for (int i = 0; i < DEPTH; ++i) {
        // residual update + layernorm
        k_resid_ln<<<NTOK, 128, 0, stream>>>(
            (i == 0) ? tokens : hidden, residual, hs,
            norm_w + i * D_MODEL, norm_b + i * D_MODEL, (i == 0) ? 0 : 1);

        // in_proj: (2352 x 384) @ (384 x 1536)
        {
            const int Ks = D_MODEL / 32;                       // 12
            k_stageA<<<cd(MTILES * Ks * FR, T), T, 0, stream>>>(
                hs, D_MODEL, NTOK, D_MODEL, sA, Ks, MTILES * Ks * FR);
            k_stageB<<<cd((2*D_INNER/16) * Ks * FR, T), T, 0, stream>>>(
                in_w + (size_t)i * D_MODEL * 2 * D_INNER, 2 * D_INNER,
                D_MODEL, 2 * D_INNER, 0, sB, Ks, (2*D_INNER/16) * Ks * FR);
            k_wmma_gemm<<<dim3(2 * D_INNER / 64, MBLKS), 32, 0, stream>>>(
                sA, sB, xz, Ks, 2 * D_INNER);
        }

        // conv + silu
        k_conv_silu<<<cd(NTOK * D_INNER, T), T, 0, stream>>>(
            xz, conv_w + (size_t)i * D_INNER * D_CONV, conv_b + i * D_INNER, xc);

        // x_proj: (2352 x 768) @ (768 x 56 -> padded 64)
        {
            const int Ks = D_INNER / 32;                       // 24
            k_stageA<<<cd(MTILES * Ks * FR, T), T, 0, stream>>>(
                xc, D_INNER, NTOK, D_INNER, sA, Ks, MTILES * Ks * FR);
            k_stageB<<<cd(4 * Ks * FR, T), T, 0, stream>>>(
                x_w + (size_t)i * D_INNER * (DT_RANK + 2 * D_STATE),
                DT_RANK + 2 * D_STATE, D_INNER, DT_RANK + 2 * D_STATE,
                0, sB, Ks, 4 * Ks * FR);
            k_wmma_gemm<<<dim3(1, MBLKS), 32, 0, stream>>>(sA, sB, dbc, Ks, 64);
        }

        // dt_proj: (2352 x 24 -> K padded 32) @ (24 -> 32 x 768)
        {
            const int Ks = 1;
            k_stageA<<<cd(MTILES * Ks * FR, T), T, 0, stream>>>(
                dbc, 64, NTOK, DT_RANK, sA, Ks, MTILES * Ks * FR);
            k_stageB<<<cd((D_INNER/16) * Ks * FR, T), T, 0, stream>>>(
                dt_w + (size_t)i * DT_RANK * D_INNER, D_INNER,
                DT_RANK, D_INNER, 0, sB, Ks, (D_INNER/16) * Ks * FR);
            k_wmma_gemm<<<dim3(D_INNER / 64, MBLKS), 32, 0, stream>>>(
                sA, sB, dtbuf, Ks, D_INNER);
        }
        k_softplus<<<cd(NTOK * D_INNER, T), T, 0, stream>>>(dtbuf, dt_b + i * D_INNER);

        // selective scan
        k_scan<<<cd(BATCH * D_INNER, T), T, 0, stream>>>(
            dtbuf, xc, dbc, A_log + (size_t)i * D_INNER * D_STATE,
            D_skip + i * D_INNER, ybuf);

        // gate + out_proj: (2352 x 768) @ (768 x 384)
        {
            const int Ks = D_INNER / 32;                       // 24
            k_gateA<<<cd(MTILES * Ks * FR, T), T, 0, stream>>>(ybuf, xz, sA);
            k_stageB<<<cd((D_MODEL/16) * Ks * FR, T), T, 0, stream>>>(
                out_w + (size_t)i * D_INNER * D_MODEL, D_MODEL,
                D_INNER, D_MODEL, 0, sB, Ks, (D_MODEL/16) * Ks * FR);
            k_wmma_gemm<<<dim3(D_MODEL / 64, MBLKS), 32, 0, stream>>>(
                sA, sB, hidden, Ks, D_MODEL);
        }
    }

    // ---- final residual + layernorm -> output ----
    k_resid_ln<<<NTOK, 128, 0, stream>>>(hidden, residual, out, normf_w, normf_b, 1);
}
